// GlmImageVQVAE_15384572854610
// MI455X (gfx1250) — compile-verified, matrix operand order checked by reading, not run on hardware
//
#include <hip/hip_runtime.h>
#include <hip/hip_bf16.h>

typedef __bf16 v16bf __attribute__((ext_vector_type(16)));
typedef float  v8f   __attribute__((ext_vector_type(8)));

#define C_DIM 256
#define D_DIM 2048
#define K_DIM 16384
#define N_TOK 4096      // B*H*W
#define HW_SZ 1024
#define BATCH 4
#define M_TILE 32       // tokens per block in the WMMA kernel
#define KSPLIT 2        // blocks splitting the K (codebook) dimension
#define KT_TOT (K_DIM / 16)
#define KT_PER (KT_TOT / KSPLIT)     // 512 column tiles per split
#define KTW    (KT_PER / 8)          // 64 column tiles per wave
#define DCHUNK 256                   // d-elements staged per B buffer
#define NCHUNK (D_DIM / DCHUNK)      // 8 chunks per column tile
#define ITEMS  (KTW * NCHUNK)        // 512 flat pipeline stages per wave
#define A_BYTES (M_TILE * D_DIM * 2)         // 128 KB
#define BBUF_BYTES (16 * DCHUNK * 2)         // 8 KB per wave per parity
#define LDS_BYTES (A_BYTES + 8 * 2 * BBUF_BYTES)  // 256 KB

union Frag  { v16bf v; uint4 u[2]; };
union Pack8 { __bf16 h[8]; uint4 u; };

// XOR-swizzle low 4 bits of the 16B-chunk index by row -> conflict-free
// ds_load_b128 across the 16 lanes of each wave half.
__device__ __forceinline__ unsigned swz(unsigned c, unsigned m) {
  return (c & ~15u) | ((c ^ m) & 15u);
}
// Same idea for the 32-chunk B columns (keep bit4, swizzle low 4 by column).
__device__ __forceinline__ unsigned bswz(unsigned c, unsigned col) {
  return (c & 16u) | ((c ^ col) & 15u);
}

// CDNA5 async global->LDS copy (ASYNCcnt-tracked, per-lane LDS dest VGPR).
__device__ __forceinline__ void async_load_b128(unsigned ldsAddr,
                                                unsigned long long gaddr) {
  asm volatile("global_load_async_to_lds_b128 %0, %1, off"
               :: "v"(ldsAddr), "v"(gaddr) : "memory");
}
__device__ __forceinline__ void wait_async0()  {
  asm volatile("s_wait_asynccnt 0x0" ::: "memory");
}
__device__ __forceinline__ void wait_async16() {
  asm volatile("s_wait_asynccnt 0x10" ::: "memory");
}

// Stage one wave-private B buffer: 16 codebook columns x DCHUNK d-elements.
// Lane 'lane' supplies chunk 'lane' (16B) of each of the 16 columns.
__device__ __forceinline__ void stage_b(const __bf16* __restrict__ cbb,
                                        unsigned ldsDst, int rowBase, int ch,
                                        int lane) {
  unsigned long long g0 = (unsigned long long)(size_t)cbb +
                          (unsigned long long)(unsigned)rowBase * 4096ull +
                          (unsigned long long)(unsigned)(ch * (DCHUNK * 2) + lane * 16);
#pragma unroll
  for (int q = 0; q < 16; ++q) {
    unsigned dst = ldsDst + ((unsigned)(q * 32) + bswz((unsigned)lane, (unsigned)q)) * 16u;
    async_load_b128(dst, g0 + (unsigned long long)q * 4096ull);
  }
}

// ---------------------------------------------------------------------------
// Kernel A: 1x1 conv (C->D per token) + L2 normalize, emit bf16 tokens.
// ---------------------------------------------------------------------------
__global__ void __launch_bounds__(256) qconv_norm_kernel(
    const float* __restrict__ hs, const float* __restrict__ qw,
    const float* __restrict__ qb, __bf16* __restrict__ hn)
{
  __shared__ float tok[C_DIM];
  __shared__ float red[256];
  int t = threadIdx.x;
  int token = blockIdx.x;
  int b = token >> 10, hw = token & 1023;
  tok[t] = hs[((size_t)b * C_DIM + t) * HW_SZ + hw];
  __syncthreads();

  int d0 = t * 8;
  float acc[8];
  float ss = 0.f;
#pragma unroll
  for (int j = 0; j < 8; ++j) {
    const float4* row = (const float4*)(qw + (size_t)(d0 + j) * C_DIM);
    float a = qb[d0 + j];
#pragma unroll 8
    for (int c4 = 0; c4 < C_DIM / 4; ++c4) {
      float4 wv = row[c4];
      a += tok[c4 * 4 + 0] * wv.x + tok[c4 * 4 + 1] * wv.y +
           tok[c4 * 4 + 2] * wv.z + tok[c4 * 4 + 3] * wv.w;
    }
    acc[j] = a;
    ss += a * a;
  }
  red[t] = ss;
  __syncthreads();
  for (int s = 128; s > 0; s >>= 1) {
    if (t < s) red[t] += red[t + s];
    __syncthreads();
  }
  float invn = 1.0f / fmaxf(sqrtf(red[0]), 1e-12f);
  Pack8 p;
#pragma unroll
  for (int j = 0; j < 8; ++j) p.h[j] = (__bf16)(acc[j] * invn);
  *(uint4*)(hn + (size_t)token * D_DIM + d0) = p.u;
}

// ---------------------------------------------------------------------------
// Kernel B: codebook -> bf16 copy + per-row 1/max(||row||, eps).
// ---------------------------------------------------------------------------
__global__ void __launch_bounds__(256) cbnorm_kernel(
    const float* __restrict__ cb, __bf16* __restrict__ cbb,
    float* __restrict__ invn)
{
  __shared__ float red[256];
  int t = threadIdx.x;
  size_t base = (size_t)blockIdx.x * D_DIM;
  int d0 = t * 8;
  const float4* src = (const float4*)(cb + base + d0);
  float4 a = src[0], b = src[1];
  float ss = a.x * a.x + a.y * a.y + a.z * a.z + a.w * a.w +
             b.x * b.x + b.y * b.y + b.z * b.z + b.w * b.w;
  Pack8 p;
  p.h[0] = (__bf16)a.x; p.h[1] = (__bf16)a.y; p.h[2] = (__bf16)a.z; p.h[3] = (__bf16)a.w;
  p.h[4] = (__bf16)b.x; p.h[5] = (__bf16)b.y; p.h[6] = (__bf16)b.z; p.h[7] = (__bf16)b.w;
  *(uint4*)(cbb + base + d0) = p.u;
  red[t] = ss;
  __syncthreads();
  for (int s = 128; s > 0; s >>= 1) {
    if (t < s) red[t] += red[t + s];
    __syncthreads();
  }
  if (t == 0) invn[blockIdx.x] = 1.0f / fmaxf(sqrtf(red[0]), 1e-12f);
}

// ---------------------------------------------------------------------------
// Kernel C: fused bf16-WMMA similarity GEMM + per-token argmax over a K-split.
// Block = 32 tokens; 8 waves. The A tile (32x2048 bf16, 128 KB) is async-
// filled into swizzled LDS once. Each wave walks 64 column tiles; its B
// operands stream through a wave-private double-buffered LDS region (2x8 KB)
// filled by global_load_async_to_lds_b128 one d-chunk ahead of the WMMAs.
// Wave-private buffers => only s_wait_asynccnt in the hot loop, no barriers,
// no VGPR software pipelining (nothing for the register allocator to spill).
// ---------------------------------------------------------------------------
__global__ void __launch_bounds__(256) argmax_wmma_kernel(
    const __bf16* __restrict__ hn, const __bf16* __restrict__ cbb,
    const float* __restrict__ invn, float* __restrict__ pv,
    int* __restrict__ pi)
{
  extern __shared__ __align__(16) char smem[];   // 256 KB (A tile + B buffers)
  int t    = threadIdx.x;
  int wave = t >> 5;
  int lane = t & 31;
  unsigned half = (unsigned)(lane >> 4);
  int l15  = lane & 15;
  int tokBase = blockIdx.x * M_TILE;
  int split   = blockIdx.y;
  int ktBase  = split * KT_PER;

  unsigned ldsBase = (unsigned)(size_t)smem;

  // Async swizzled fill of the A tile: 8192 chunks of 16B, 32 per thread.
  {
    unsigned long long gbase =
        (unsigned long long)(size_t)(hn + (size_t)tokBase * D_DIM);
#pragma unroll
    for (int i = 0; i < 32; ++i) {
      int cid = t + i * 256;               // 0..8191
      int m = cid >> 8;                    // row 0..31
      int c = cid & 255;                   // chunk 0..255
      unsigned dst = ldsBase + (unsigned)((m * 256 + swz((unsigned)c, (unsigned)m)) * 16);
      async_load_b128(dst, gbase + (unsigned long long)cid * 16ull);
    }
    wait_async0();
  }
  __syncthreads();

  const uint4* a4 = (const uint4*)smem;
  unsigned bWave = ldsBase + (unsigned)A_BYTES + (unsigned)(wave * 2 * BBUF_BYTES);

  float bv0[8], bv1[8];
  int   bi0[8], bi1[8];
#pragma unroll
  for (int j = 0; j < 8; ++j) {
    bv0[j] = -3.402823466e38f; bi0[j] = 0;
    bv1[j] = -3.402823466e38f; bi1[j] = 0;
  }

  unsigned m0 = (unsigned)l15;        // A row, sub-tile 0 (tokens 0..15)
  unsigned m1 = (unsigned)(l15 + 16); // sub-tile 1 (tokens 16..31)

  v8f acc0 = {}, acc1 = {};

  // Prologue: stage first B chunk (ktw=0, ch=0) into buffer parity 0.
  stage_b(cbb, bWave, (ktBase + wave) * 16, 0, lane);

#pragma unroll 1
  for (int it = 0; it < ITEMS; ++it) {
    int parity = it & 1;
    int ktw = it >> 3;                 // wave-local column-tile index
    int ch  = it & 7;                  // d-chunk within the column tile

    if (it + 1 < ITEMS) {
      int itN  = it + 1;
      int ktwN = itN >> 3;
      int chN  = itN & 7;
      int rowBaseN = (ktBase + ktwN * 8 + wave) * 16;
      stage_b(cbb, bWave + (unsigned)((itN & 1) * BBUF_BYTES), rowBaseN, chN, lane);
      wait_async16();                  // stage 'it' done; stage 'it+1' in flight
    } else {
      wait_async0();
    }

    if (ch == 0) { acc0 = (v8f){}; acc1 = (v8f){}; }

    const uint4* bb = (const uint4*)(smem + A_BYTES +
                                     wave * 2 * BBUF_BYTES + parity * BBUF_BYTES);
#pragma unroll
    for (int js = 0; js < 8; ++js) {
      unsigned cA = (unsigned)(ch * 32 + js * 4);   // A-tile 16B-chunk base
      unsigned cr = (unsigned)(js * 4);             // B-buffer chunk base
      Frag a0, a1, b;
      b.u[0]  = bb[l15 * 32 + bswz(cr + half, (unsigned)l15)];
      b.u[1]  = bb[l15 * 32 + bswz(cr + 2 + half, (unsigned)l15)];
      a0.u[0] = a4[m0 * 256 + swz(cA + half, m0)];
      a0.u[1] = a4[m0 * 256 + swz(cA + 2 + half, m0)];
      a1.u[0] = a4[m1 * 256 + swz(cA + half, m1)];
      a1.u[1] = a4[m1 * 256 + swz(cA + 2 + half, m1)];
      acc0 = __builtin_amdgcn_wmma_f32_16x16x32_bf16(false, a0.v, false, b.v,
                                                     (short)0, acc0, false, false);
      acc1 = __builtin_amdgcn_wmma_f32_16x16x32_bf16(false, a1.v, false, b.v,
                                                     (short)0, acc1, false, false);
    }

    if (ch == 7) {                     // full-D dot done for this column tile
      int r = (ktBase + ktw * 8 + wave) * 16 + l15;
      float sc = invn[r];
#pragma unroll
      for (int j = 0; j < 8; ++j) {
        float v0 = acc0[j] * sc;
        float v1 = acc1[j] * sc;
        if (v0 > bv0[j]) { bv0[j] = v0; bi0[j] = r; }  // ascending k -> first max
        if (v1 > bv1[j]) { bv1[j] = v1; bi1[j] = r; }
      }
    }
  }

  // Reduce across the 16 lanes of each half (same row set per half).
#pragma unroll
  for (int j = 0; j < 8; ++j) {
#pragma unroll
    for (int off = 1; off < 16; off <<= 1) {
      float ov; int oi;
      ov = __shfl_xor(bv0[j], off, 32); oi = __shfl_xor(bi0[j], off, 32);
      if (ov > bv0[j] || (ov == bv0[j] && oi < bi0[j])) { bv0[j] = ov; bi0[j] = oi; }
      ov = __shfl_xor(bv1[j], off, 32); oi = __shfl_xor(bi1[j], off, 32);
      if (ov > bv1[j] || (ov == bv1[j] && oi < bi1[j])) { bv1[j] = ov; bi1[j] = oi; }
    }
  }

  __syncthreads();                    // A tile dead; reuse as cross-wave scratch
  float* rv = (float*)smem;           // [32 rows][8 waves]
  int*   ri = (int*)smem + 256;
  if (l15 == 0) {
#pragma unroll
    for (int j = 0; j < 8; ++j) {
      int row = j + ((int)half << 3);
      rv[row * 8 + wave]        = bv0[j];
      ri[row * 8 + wave]        = bi0[j];
      rv[(row + 16) * 8 + wave] = bv1[j];
      ri[(row + 16) * 8 + wave] = bi1[j];
    }
  }
  __syncthreads();
  if (t < M_TILE) {
    float best = -3.402823466e38f;
    int bidx = 0x7fffffff;
    for (int w = 0; w < 8; ++w) {
      float v = rv[t * 8 + w];
      int   i = ri[t * 8 + w];
      if (v > best || (v == best && i < bidx)) { best = v; bidx = i; }
    }
    pv[split * N_TOK + tokBase + t] = best;
    pi[split * N_TOK + tokBase + t] = bidx;
  }
}

// ---------------------------------------------------------------------------
// Merge the KSPLIT partial argmax results (tie -> lower codebook index).
// ---------------------------------------------------------------------------
__global__ void __launch_bounds__(256) merge_kernel(
    const float* __restrict__ pv, const int* __restrict__ pi,
    int* __restrict__ idx_out)
{
  int tok = blockIdx.x * 256 + threadIdx.x;
  float bv = pv[tok];
  int   bi = pi[tok];
#pragma unroll
  for (int sp = 1; sp < KSPLIT; ++sp) {
    float v = pv[sp * N_TOK + tok];
    int   i = pi[sp * N_TOK + tok];
    if (v > bv || (v == bv && i < bi)) { bv = v; bi = i; }
  }
  idx_out[tok] = bi;
}

// ---------------------------------------------------------------------------
// Kernel D: quant[b,d,h,w] = codebook[idx[token], d] * invnorm[idx[token]].
// One thread per output element -> perfectly coalesced stores.
// ---------------------------------------------------------------------------
__global__ void __launch_bounds__(256) gather_kernel(
    const float* __restrict__ cb, const float* __restrict__ invn,
    const int* __restrict__ idx, float* __restrict__ out)
{
  int o  = blockIdx.x * 256 + threadIdx.x;   // 0 .. 8388607
  int hw = o & 1023;
  int d  = (o >> 10) & 2047;
  int b  = o >> 21;
  int token = b * HW_SZ + hw;
  int k = idx[token];
  out[o] = cb[(size_t)k * D_DIM + d] * invn[k];
}

extern "C" void kernel_launch(void* const* d_in, const int* in_sizes, int n_in,
                              void* d_out, int out_size, void* d_ws, size_t ws_size,
                              hipStream_t stream) {
  const float* hs = (const float*)d_in[0];  // (4,256,32,32)
  const float* cb = (const float*)d_in[1];  // (16384,2048)
  const float* qw = (const float*)d_in[2];  // (2048,256)
  const float* qb = (const float*)d_in[3];  // (2048,)

  char* ws = (char*)d_ws;
  size_t off = 0;
  __bf16* cbb = (__bf16*)(ws + off); off += (size_t)K_DIM * D_DIM * 2;  // 64 MiB
  __bf16* hn  = (__bf16*)(ws + off); off += (size_t)N_TOK * D_DIM * 2;  // 16 MiB
  float* invn = (float*)(ws + off);  off += (size_t)K_DIM * 4;          // 64 KiB
  float* pv   = (float*)(ws + off);  off += (size_t)KSPLIT * N_TOK * 4; // 32 KiB
  int*   pi   = (int*)(ws + off);    off += (size_t)KSPLIT * N_TOK * 4; // 32 KiB

  float* quant_out = (float*)d_out;                                  // 8388608 f32
  int* idx_out = (int*)(quant_out + (size_t)BATCH * D_DIM * HW_SZ);  // 4096 i32

  qconv_norm_kernel<<<N_TOK, 256, 0, stream>>>(hs, qw, qb, hn);
  cbnorm_kernel<<<K_DIM, 256, 0, stream>>>(cb, cbb, invn);
  dim3 cgrid(N_TOK / M_TILE, KSPLIT);
  argmax_wmma_kernel<<<cgrid, 256, LDS_BYTES, stream>>>(hn, cbb, invn, pv, pi);
  merge_kernel<<<N_TOK / 256, 256, 0, stream>>>(pv, pi, idx_out);
  gather_kernel<<<(BATCH * D_DIM * HW_SZ) / 256, 256, 0, stream>>>(cb, invn,
                                                                   idx_out, quant_out);
}
